// CrossAttention_87376814669840
// MI455X (gfx1250) — matrix operand order repaired and checked by reading, not compiled, not run on hardware
//
#include <hip/hip_runtime.h>
#include <hip/hip_bf16.h>
#include <math.h>

typedef _Float16 h16;
typedef __attribute__((ext_vector_type(16))) _Float16 v16h;
typedef __attribute__((ext_vector_type(8)))  _Float16 v8h;
typedef __attribute__((ext_vector_type(2)))  _Float16 v2h;
typedef __attribute__((ext_vector_type(8)))  float    v8f;
typedef __attribute__((ext_vector_type(4)))  unsigned int v4u;
typedef __attribute__((ext_vector_type(8)))  int      v8i;
typedef __attribute__((ext_vector_type(4)))  int      v4i;

#define DIM 768
#define HEADS 12
#define HD 64

static __device__ __forceinline__ v8f wmma_f16(v16h a, v16h b, v8f c) {
  return __builtin_amdgcn_wmma_f32_16x16x32_f16(false, a, false, b, (short)0, c, false, false);
}

static __device__ __forceinline__ v16h cat16(v8h lo, v8h hi) {
  return __builtin_shufflevector(lo, hi, 0, 1, 2, 3, 4, 5, 6, 7, 8, 9, 10, 11, 12, 13, 14, 15);
}

// ---------------------------------------------------------------------------
// TDM 2D tile load: global f16 tensor (row stride `stride` elems) -> LDS,
// with LDS row padding (pad_interval/pad_amount codes per ISA 8.4).
// D# group0/group1 built per CDNA5 ISA §8.3/8.4; groups 2/3 zero (2-D).
// This toolchain exposes the 6-arg builtin (g0, g1, g2, g3, g_extra, cpol).
// ---------------------------------------------------------------------------
static __device__ __forceinline__ void tdm_load_2d(
    unsigned lds_addr, const void* gaddr,
    unsigned d0len, unsigned d1len,      // tensor dims (elements, for OOB)
    unsigned t0, unsigned t1,            // tile dims (elements)
    unsigned stride,                     // dim0 stride (elements)
    unsigned padiv, unsigned padamt) {   // pad codes (3->16DW, 4->32DW; amt 3->4DW)
  unsigned long long ga = (unsigned long long)gaddr;
  v4u g0;
  g0.x = 1u;                                       // count=1, user mode
  g0.y = lds_addr;                                 // lds_addr (bytes)
  g0.z = (unsigned)ga;                             // global_addr lo
  g0.w = (unsigned)((ga >> 32) & 0x1FFFFFFu) | (2u << 30);  // addr hi | type=2
  v8i g1;
  g1[0] = (int)((1u << 16) | (1u << 20) | (padiv << 22) | (padamt << 25)); // 2B data, pad_enable
  g1[1] = (int)((d0len & 0xFFFFu) << 16);          // tensor_dim0[15:0] (atomic addr = 0)
  g1[2] = (int)(((d0len >> 16) & 0xFFFFu) | ((d1len & 0xFFFFu) << 16));
  g1[3] = (int)(((d1len >> 16) & 0xFFFFu) | ((t0 & 0xFFFFu) << 16));      // | tile_dim0
  g1[4] = (int)(t1 & 0xFFFFu);                     // tile_dim1 (tile_dim2=0)
  g1[5] = (int)stride;                             // tensor_dim0_stride lo32
  g1[6] = 0;                                       // stride hi16 | dim1_stride lo16
  g1[7] = 0;
  v4i z4 = (v4i){0, 0, 0, 0};
  v8i z8 = (v8i){0, 0, 0, 0, 0, 0, 0, 0};
  __builtin_amdgcn_tensor_load_to_lds(g0, g1, z4, z4, z8, 0);
}

// ---------------------------------------------------------------------------
// Weight convert+transpose: W f32 [K][N] -> Wt f16 [N][K] (32x32 LDS tiles).
// ---------------------------------------------------------------------------
__global__ __launch_bounds__(256) void wtr_kernel(const float* __restrict__ in,
                                                  h16* __restrict__ out, int K, int N) {
  __shared__ h16 t[32][33];
  int k0 = blockIdx.y * 32, n0 = blockIdx.x * 32;
  int lane = threadIdx.x & 31, w8 = threadIdx.x >> 5;
#pragma unroll
  for (int it = 0; it < 4; ++it) {
    int r = w8 + it * 8;
    t[r][lane] = (h16)in[(size_t)(k0 + r) * N + n0 + lane];
  }
  __syncthreads();
#pragma unroll
  for (int it = 0; it < 4; ++it) {
    int r = w8 + it * 8;
    out[(size_t)(n0 + r) * K + k0 + lane] = t[lane][r];
  }
}

// ---------------------------------------------------------------------------
// LayerNorm: f32 in -> f16 out. One wave per row of 768, 8 rows per block.
// ---------------------------------------------------------------------------
__global__ __launch_bounds__(256) void ln_kernel(const float* __restrict__ in,
                                                 h16* __restrict__ out,
                                                 const float* __restrict__ gam,
                                                 const float* __restrict__ bet) {
  int row  = blockIdx.x * 8 + (threadIdx.x >> 5);
  int lane = threadIdx.x & 31;
  const float* p = in + (size_t)row * DIM;
  float vals[DIM / 32];
  float s = 0.f;
#pragma unroll
  for (int i = 0; i < DIM / 32; ++i) { vals[i] = p[lane + i * 32]; s += vals[i]; }
#pragma unroll
  for (int off = 16; off > 0; off >>= 1) s += __shfl_xor(s, off, 32);
  float mean = s * (1.0f / DIM);
  float v = 0.f;
#pragma unroll
  for (int i = 0; i < DIM / 32; ++i) { float d = vals[i] - mean; v += d * d; }
#pragma unroll
  for (int off = 16; off > 0; off >>= 1) v += __shfl_xor(v, off, 32);
  float inv = rsqrtf(v * (1.0f / DIM) + 1e-5f);
  h16* o = out + (size_t)row * DIM;
#pragma unroll
  for (int i = 0; i < DIM / 32; ++i) {
    int c = lane + i * 32;
    o[c] = (h16)((vals[i] - mean) * inv * gam[c] + bet[c]);
  }
}

// ---------------------------------------------------------------------------
// WMMA GEMM, all-f16 operands staged by TDM with double buffering.
// C(MxN) = epi(A(MxK,f16) * Bt(NxK,f16)^T + bias)   out: f16 or f32(+res)
// Block = 8 waves, tile 32x256. LDS rows padded 64B->80B by TDM pad feature.
// ---------------------------------------------------------------------------
enum { EPI_BIAS = 0, EPI_GELU = 1, EPI_RES = 2 };

template <int EPI, bool OUTH>
__global__ __launch_bounds__(256) void gemm_kernel(
    const h16* __restrict__ A, const h16* __restrict__ Bt,
    const float* __restrict__ bias, const float* __restrict__ res,
    void* __restrict__ Cout, int M, int N, int K) {
  __shared__ h16 As[2][32][40];    // [m][k], 80B rows (TDM-padded)
  __shared__ h16 BsT[2][256][40];  // [n][k], 80B rows (TDM-padded)
  int tid  = threadIdx.x;
  int m0   = blockIdx.y * 32;
  int n0   = blockIdx.x * 256;
  int lane = tid & 31;
  int wave = tid >> 5;
  int wm   = wave >> 2;
  int wn   = wave & 3;
  v8f acc[4];
#pragma unroll
  for (int t = 0; t < 4; ++t) acc[t] = (v8f){};

  int nsteps = K >> 5;
  if (wave == 0) {
    tdm_load_2d((unsigned)(size_t)&As[0][0][0], A + (size_t)m0 * K, K, M, 32, 32, K, 3, 3);
    tdm_load_2d((unsigned)(size_t)&BsT[0][0][0], Bt + (size_t)n0 * K, K, N, 32, 256, K, 3, 3);
  }
  for (int step = 0; step < nsteps; ++step) {
    int cur = step & 1;
    if (wave == 0) __builtin_amdgcn_s_wait_tensorcnt(0);  // buffer[cur] landed
    __syncthreads();
    if (wave == 0 && step + 1 < nsteps) {                 // prefetch next tiles
      int k1 = (step + 1) << 5;
      tdm_load_2d((unsigned)(size_t)&As[cur ^ 1][0][0], A + (size_t)m0 * K + k1, K, M, 32, 32, K, 3, 3);
      tdm_load_2d((unsigned)(size_t)&BsT[cur ^ 1][0][0], Bt + (size_t)n0 * K + k1, K, N, 32, 256, K, 3, 3);
    }
    int mr = 16 * wm + (lane & 15);
    int ka = 8 * (lane >> 4);
    v16h a = cat16(*(const v8h*)&As[cur][mr][ka], *(const v8h*)&As[cur][mr][ka + 16]);
    int kb = 8 * (lane >> 4);
#pragma unroll
    for (int t = 0; t < 4; ++t) {
      int nn = 64 * wn + 16 * t + (lane & 15);
      v16h b = cat16(*(const v8h*)&BsT[cur][nn][kb], *(const v8h*)&BsT[cur][nn][kb + 16]);
      acc[t] = wmma_f16(a, b, acc[t]);
    }
    __syncthreads();
  }

#pragma unroll
  for (int t = 0; t < 4; ++t) {
#pragma unroll
    for (int j = 0; j < 8; ++j) {
      int row = m0 + 16 * wm + j + 8 * (lane >> 4);
      int col = n0 + 64 * wn + 16 * t + (lane & 15);
      float v = acc[t][j] + bias[col];
      if (EPI == EPI_GELU) v = 0.5f * v * (1.0f + erff(v * 0.70710678118f));
      if (EPI == EPI_RES)  v += res[(size_t)row * N + col];
      if (OUTH) ((h16*)Cout)[(size_t)row * N + col] = (h16)v;
      else      ((float*)Cout)[(size_t)row * N + col] = v;
    }
  }
}

// ---------------------------------------------------------------------------
// Grouped 3x3 conv (4 ch/group), fused: out = (in + conv(in) + bias) * scale
// f16 in/out. in: (B*H*W, ldin) token rows; out: (B*H*W, 768)
// ---------------------------------------------------------------------------
__global__ __launch_bounds__(256) void gconv_kernel(
    const h16* __restrict__ in, int ldin,
    const float* __restrict__ wt, const float* __restrict__ bias,
    h16* __restrict__ out, int H, int W, float scale, int total) {
  int idx = blockIdx.x * 256 + threadIdx.x;
  if (idx >= total) return;
  int xw = idx % W;
  int t  = idx / W;
  int y  = t % H; t /= H;
  int o  = t % DIM;
  int b  = t / DIM;
  int HW = H * W;
  int gbase = o & ~3;
  float acc = bias[o];
#pragma unroll
  for (int i = 0; i < 4; ++i) {
    int ci = gbase + i;
    const float* wp = wt + (size_t)(o * 4 + i) * 9;
#pragma unroll
    for (int dy = 0; dy < 3; ++dy) {
      int yy = y + dy - 1;
      if (yy < 0 || yy >= H) continue;
#pragma unroll
      for (int dx = 0; dx < 3; ++dx) {
        int xx = xw + dx - 1;
        if (xx < 0 || xx >= W) continue;
        acc += (float)in[(size_t)(b * HW + yy * W + xx) * ldin + ci] * wp[dy * 3 + dx];
      }
    }
  }
  float q = (float)in[(size_t)(b * HW + y * W + xw) * ldin + o];
  out[(size_t)(b * HW + y * W + xw) * DIM + o] = (h16)((q + acc) * scale);
}

// ---------------------------------------------------------------------------
// Attention per (batch, head): S = softmax(Q K^T), O = S V.  All-f16 I/O.
// Q/K staged by TDM (128B rows -> 144B padded LDS rows); V transposed by VALU.
// S f32 in LDS, narrowed in place to f16 P after softmax.
// ---------------------------------------------------------------------------
template <int LQ, int LK>
__global__ __launch_bounds__(256) void attn_kernel(
    const h16* __restrict__ Qg, const h16* __restrict__ Kg,
    const h16* __restrict__ Vg, int ldv, h16* __restrict__ Og) {
  __shared__ h16 Qs[LQ][HD + 8];    // 144B rows (TDM-padded)
  __shared__ h16 Ks[LK][HD + 8];    // 144B rows (TDM-padded)
  __shared__ h16 VsT[HD][LK + 8];   // [d][k]
  __shared__ float S[LQ][LK + 4];   // 16B-aligned rows
  int tid = threadIdx.x;
  int bh  = blockIdx.x;
  int b   = bh / HEADS, h = bh % HEADS;
  const h16* qp = Qg + (size_t)(b * LQ) * DIM + h * HD;
  const h16* kp = Kg + (size_t)(b * LK) * DIM + h * HD;
  const h16* vp = Vg + (size_t)(b * LK) * ldv + h * HD;
  int lane = tid & 31, wave = tid >> 5;
  if (wave == 0) {
    tdm_load_2d((unsigned)(size_t)&Qs[0][0], qp, DIM, LQ, HD, LQ, DIM, 4, 3);
    tdm_load_2d((unsigned)(size_t)&Ks[0][0], kp, DIM, LK, HD, LK, DIM, 4, 3);
  }
  for (int i = tid; i < HD * (LK / 2); i += 256) {
    int c = i & 63, r2 = i >> 6;
    const h16* p = vp + (size_t)(2 * r2) * ldv + c;
    v2h val = {p[0], p[ldv]};
    *(v2h*)&VsT[c][2 * r2] = val;
  }
  if (wave == 0) __builtin_amdgcn_s_wait_tensorcnt(0);
  __syncthreads();

  // --- S = Q * K^T ---
  constexpr int TQ = LQ / 16, TK = LK / 16;
  for (int t = wave; t < TQ * TK; t += 8) {
    int tm = t / TK, tn = t % TK;
    v8f c = (v8f){};
#pragma unroll
    for (int k0 = 0; k0 < HD; k0 += 32) {
      int mr = tm * 16 + (lane & 15);
      int ka = k0 + 8 * (lane >> 4);
      v16h a = cat16(*(const v8h*)&Qs[mr][ka], *(const v8h*)&Qs[mr][ka + 16]);
      int nr = tn * 16 + (lane & 15);
      int kb = k0 + 8 * (lane >> 4);
      v16h bb = cat16(*(const v8h*)&Ks[nr][kb], *(const v8h*)&Ks[nr][kb + 16]);
      c = wmma_f16(a, bb, c);
    }
#pragma unroll
    for (int j = 0; j < 8; ++j)
      S[tm * 16 + j + 8 * (lane >> 4)][tn * 16 + (lane & 15)] = c[j];
  }
  __syncthreads();

  // --- row softmax (f32), then narrow P to f16 in place ---
  for (int r = tid; r < LQ; r += 256) {
    float mx = -1e30f;
    for (int k = 0; k < LK; ++k) mx = fmaxf(mx, S[r][k]);
    float sum = 0.f;
    for (int k = 0; k < LK; ++k) { float e = expf(S[r][k] - mx); S[r][k] = e; sum += e; }
    float inv = 1.0f / sum;
    h16* prow = (h16*)&S[r][0];
    for (int k = 0; k < LK; ++k) prow[k] = (h16)(S[r][k] * inv);
  }
  __syncthreads();

  // --- O = P * V ---
  h16* op = Og + (size_t)(b * LQ) * DIM + h * HD;
  for (int t = wave; t < TQ * 4; t += 8) {
    int tm = t >> 2, tn = t & 3;
    v8f c = (v8f){};
#pragma unroll
    for (int k0 = 0; k0 < LK; k0 += 32) {
      int mr = tm * 16 + (lane & 15);
      const h16* prow = (const h16*)&S[mr][0];
      int ka = k0 + 8 * (lane >> 4);
      v16h a = cat16(*(const v8h*)&prow[ka], *(const v8h*)&prow[ka + 16]);
      int nn = tn * 16 + (lane & 15);
      int kb = k0 + 8 * (lane >> 4);
      v16h bb = cat16(*(const v8h*)&VsT[nn][kb], *(const v8h*)&VsT[nn][kb + 16]);
      c = wmma_f16(a, bb, c);
    }
#pragma unroll
    for (int j = 0; j < 8; ++j)
      op[(size_t)(tm * 16 + j + 8 * (lane >> 4)) * DIM + tn * 16 + (lane & 15)] = (h16)c[j];
  }
}

// ---------------------------------------------------------------------------
// Host orchestration
// ---------------------------------------------------------------------------
extern "C" void kernel_launch(void* const* d_in, const int* in_sizes, int n_in,
                              void* d_out, int out_size, void* d_ws, size_t ws_size,
                              hipStream_t stream) {
  (void)in_sizes; (void)n_in; (void)out_size; (void)ws_size;
  auto P = [&](int i) { return (const float*)d_in[i]; };
  const float* z = P(0);
  const float* x = P(1);
  constexpr size_t NZ = 64 * 64;
  constexpr size_t NX = 64 * 256;

  float* out_z = (float*)d_out;
  float* out_x = out_z + NZ * DIM;

  // f16 workspace arena
  h16* hw   = (h16*)d_ws;
  h16* zn   = hw;                    // NZ*768
  h16* xn   = zn + NZ * DIM;         // NX*768
  h16* qraw = xn + NX * DIM;         // NX*768
  h16* kv   = qraw + NX * DIM;       // NX*1536
  h16* qsum = kv + NX * 2 * DIM;     // NX*768
  h16* ksum = qsum + NX * DIM;       // NX*768
  h16* wts  = ksum + NX * DIM;       // f16 transposed weights
  h16* qwT_zx  = wts;
  h16* kvwT_zx = qwT_zx + 768 * 768;
  h16* pwT_zx  = kvwT_zx + 768 * 1536;
  h16* qwT_xz  = pwT_zx + 768 * 768;
  h16* kvwT_xz = qwT_xz + 768 * 768;
  h16* pwT_xz  = kvwT_xz + 768 * 1536;
  h16* w1T_z   = pwT_xz + 768 * 768;
  h16* w2T_z   = w1T_z + 768 * 3072;
  h16* w1T_x   = w2T_z + 768 * 3072;
  h16* w2T_x   = w1T_x + 768 * 3072;
  h16* attnb = qraw;   // alias: qraw dead once qsum built
  h16* tbuf  = qsum;   // phase-B LN output
  h16* h1    = hw;     // phase-B hidden (NX*3072 fits below qsum)

  // -------- weight convert + transpose (f32 [K][N] -> f16 [N][K]) --------
  wtr_kernel<<<dim3(24, 24), 256, 0, stream>>>(P(10), qwT_zx, 768, 768);
  wtr_kernel<<<dim3(48, 24), 256, 0, stream>>>(P(12), kvwT_zx, 768, 1536);
  wtr_kernel<<<dim3(24, 24), 256, 0, stream>>>(P(18), pwT_zx, 768, 768);
  wtr_kernel<<<dim3(24, 24), 256, 0, stream>>>(P(20), qwT_xz, 768, 768);
  wtr_kernel<<<dim3(48, 24), 256, 0, stream>>>(P(22), kvwT_xz, 768, 1536);
  wtr_kernel<<<dim3(24, 24), 256, 0, stream>>>(P(28), pwT_xz, 768, 768);
  wtr_kernel<<<dim3(96, 24), 256, 0, stream>>>(P(30), w1T_z, 768, 3072);
  wtr_kernel<<<dim3(24, 96), 256, 0, stream>>>(P(32), w2T_z, 3072, 768);
  wtr_kernel<<<dim3(96, 24), 256, 0, stream>>>(P(34), w1T_x, 768, 3072);
  wtr_kernel<<<dim3(24, 96), 256, 0, stream>>>(P(36), w2T_x, 3072, 768);

  // -------- initial LayerNorms --------
  ln_kernel<<<NZ / 8, 256, 0, stream>>>(z, zn, P(2), P(3));
  ln_kernel<<<NX / 8, 256, 0, stream>>>(x, xn, P(4), P(5));

  // -------- z <- z + CrossAttn(zn (8x8) <- xn (16x16)) --------
  gemm_kernel<EPI_BIAS, true><<<dim3(3, NZ / 32), 256, 0, stream>>>(zn, qwT_zx, P(11), nullptr, qraw, NZ, 768, 768);
  gemm_kernel<EPI_BIAS, true><<<dim3(6, NX / 32), 256, 0, stream>>>(xn, kvwT_zx, P(13), nullptr, kv, NX, 1536, 768);
  gconv_kernel<<<(int)((NZ * DIM + 255) / 256), 256, 0, stream>>>(qraw, 768, P(14), P(15), qsum, 8, 8, 0.125f, (int)(NZ * DIM));
  gconv_kernel<<<(int)((NX * DIM + 255) / 256), 256, 0, stream>>>(kv, 1536, P(16), P(17), ksum, 16, 16, 1.0f, (int)(NX * DIM));
  attn_kernel<64, 256><<<64 * HEADS, 256, 0, stream>>>(qsum, ksum, kv + DIM, 1536, attnb);
  gemm_kernel<EPI_RES, false><<<dim3(3, NZ / 32), 256, 0, stream>>>(attnb, pwT_zx, P(19), z, out_z, NZ, 768, 768);

  // -------- x <- x + CrossAttn(xn (16x16) <- zn (8x8)) --------
  gemm_kernel<EPI_BIAS, true><<<dim3(3, NX / 32), 256, 0, stream>>>(xn, qwT_xz, P(21), nullptr, qraw, NX, 768, 768);
  gemm_kernel<EPI_BIAS, true><<<dim3(6, NZ / 32), 256, 0, stream>>>(zn, kvwT_xz, P(23), nullptr, kv, NZ, 1536, 768);
  gconv_kernel<<<(int)((NX * DIM + 255) / 256), 256, 0, stream>>>(qraw, 768, P(24), P(25), qsum, 16, 16, 0.125f, (int)(NX * DIM));
  gconv_kernel<<<(int)((NZ * DIM + 255) / 256), 256, 0, stream>>>(kv, 1536, P(26), P(27), ksum, 8, 8, 1.0f, (int)(NZ * DIM));
  attn_kernel<256, 64><<<64 * HEADS, 256, 0, stream>>>(qsum, ksum, kv + DIM, 1536, attnb);
  gemm_kernel<EPI_RES, false><<<dim3(3, NX / 32), 256, 0, stream>>>(attnb, pwT_xz, P(29), x, out_x, NX, 768, 768);

  // -------- z MLP --------
  ln_kernel<<<NZ / 8, 256, 0, stream>>>(out_z, tbuf, P(6), P(7));
  gemm_kernel<EPI_GELU, true><<<dim3(12, NZ / 32), 256, 0, stream>>>(tbuf, w1T_z, P(31), nullptr, h1, NZ, 3072, 768);
  gemm_kernel<EPI_RES, false><<<dim3(3, NZ / 32), 256, 0, stream>>>(h1, w2T_z, P(33), out_z, out_z, NZ, 768, 3072);

  // -------- x MLP --------
  ln_kernel<<<NX / 8, 256, 0, stream>>>(out_x, tbuf, P(8), P(9));
  gemm_kernel<EPI_GELU, true><<<dim3(12, NX / 32), 256, 0, stream>>>(tbuf, w1T_x, P(35), nullptr, h1, NX, 3072, 768);
  gemm_kernel<EPI_RES, false><<<dim3(3, NX / 32), 256, 0, stream>>>(h1, w2T_x, P(37), out_x, out_x, NX, 768, 3072);
}
